// Kernel_23089744183781
// MI455X (gfx1250) — matrix operand order repaired and checked by reading, not compiled, not run on hardware
//
#include <hip/hip_runtime.h>

typedef __attribute__((ext_vector_type(16))) _Float16 v16h;
typedef __attribute__((ext_vector_type(8)))  float    v8f;

#define NPTS 131072
#define KDIM 864          // 9 * 96
#define KC   27           // 864 / 32
#define PDIM 256
#define MB_HALFS (PDIM * KDIM)   // 221184 f16 elements in fragment order

// ---------------------------------------------------------------------------
// prep_mB: M1 (256 x 864, f32, row-major over (p, c=y*96+w)) -> f16 buffer in
// V_WMMA_F32_16X16X32_F16 B fragments in contiguous k order (k = 16*half + e, same as sA):
//   mB[((pt*27 + kc)*32 + lane)*16 + e] = M1[p=pt*16+(lane&15)][K=kc*32+(lane>>4)*16+e]
// so the GEMM kernel loads B with one contiguous 32B/lane vector load.
// ---------------------------------------------------------------------------
__global__ void prep_mB(const float* __restrict__ M1, _Float16* __restrict__ mB) {
    unsigned t    = blockIdx.x * blockDim.x + threadIdx.x;   // < 221184
    unsigned e    = t & 15u;
    unsigned lane = (t >> 4) & 31u;
    unsigned grp  = t >> 9;                 // pt*27 + kc
    unsigned kc   = grp % 27u;
    unsigned pt   = grp / 27u;
    unsigned p    = pt * 16u + (lane & 15u);
    unsigned K    = kc * 32u + (lane >> 4) * 16u + e;
    mB[t] = (_Float16)M1[p * 864u + K];
}

// k2[p] = sum_k M2[p][k] * weight[k]   (constant fallback 16x16 matrix)
__global__ void prep_k2(const float* __restrict__ M2, const float* __restrict__ wt,
                        float* __restrict__ k2) {
    int p = threadIdx.x;                    // 256 threads
    float acc = 0.f;
#pragma unroll
    for (int k = 0; k < 32; ++k) acc = fmaf(M2[p * 32 + k], wt[k], acc);
    k2[p] = acc;
}

// ---------------------------------------------------------------------------
// Fused main kernel: 16 points per block, 16 waves (wave32).
//   Phase A: radii / SH(Y) / h / R in LDS; F = Y (x) R written straight into
//            WMMA A-fragment layout (f16) in LDS.
//   Phase B: wave wv computes the 16x16 output tile (16 points x p-cols
//            wv*16..wv*16+15) with 27 chained v_wmma_f32_16x16x32_f16 (K=864).
// ---------------------------------------------------------------------------
__global__ __launch_bounds__(512) void sh_mix_wmma(
    const float* __restrict__ r,  const _Float16* __restrict__ mB,
    const float* __restrict__ k2, const float* __restrict__ W1,
    const float* __restrict__ b1, const float* __restrict__ W2,
    const float* __restrict__ b2, float* __restrict__ out)
{
    __shared__ __align__(32) _Float16 sA[KC * 32 * 16];  // 27 KB, A fragments
    __shared__ float sH[16 * 64];
    __shared__ float sR[16 * 96];
    __shared__ float sY[16 * 9];
    __shared__ float sRad[16];
    __shared__ int   sMask[16];

    const int t  = threadIdx.x;
    const int n0 = blockIdx.x << 4;

    // ---- Phase A1: radii + spherical harmonics (one thread per point) ----
    if (t < 16) {
        float x = r[(n0 + t) * 3 + 0];
        float y = r[(n0 + t) * 3 + 1];
        float z = r[(n0 + t) * 3 + 2];
        float rad = sqrtf(x * x + y * y + z * z);
        sRad[t]  = rad;
        sMask[t] = rad > 0.f;
        float rn = fmaxf(rad, 1e-12f);
        x /= rn; y /= rn; z /= rn;
        const float c1 = 0.4886025119029199f, c2 = 1.0925484305920792f;
        sY[t * 9 + 0] = 0.28209479177387814f;
        sY[t * 9 + 1] = c1 * y;
        sY[t * 9 + 2] = c1 * z;
        sY[t * 9 + 3] = c1 * x;
        sY[t * 9 + 4] = c2 * x * y;
        sY[t * 9 + 5] = c2 * y * z;
        sY[t * 9 + 6] = 0.31539156525252005f * (3.f * z * z - 1.f);
        sY[t * 9 + 7] = c2 * x * z;
        sY[t * 9 + 8] = 0.5462742152960396f * (x * x - y * y);
    }
    __syncthreads();

    // ---- Phase A2: h = relu(rad * W1 + b1), 16x64 values ----
#pragma unroll
    for (int q = 0; q < 2; ++q) {
        int idx = t + 512 * q;
        int pt = idx >> 6, j = idx & 63;
        sH[idx] = fmaxf(0.f, fmaf(sRad[pt], W1[j], b1[j]));
    }
    __syncthreads();

    // ---- Phase A3: R = h @ W2 + b2, 16x96 values ----
#pragma unroll
    for (int q = 0; q < 3; ++q) {
        int idx = t + 512 * q;                 // = pt*96 + w
        int pt = idx / 96, w = idx - pt * 96;
        float acc = b2[w];
#pragma unroll
        for (int j = 0; j < 64; ++j)
            acc = fmaf(sH[pt * 64 + j], W2[j * 96 + w], acc);
        sR[idx] = acc;
    }
    __syncthreads();

    // ---- Phase A4: F[pt][c] = Y[pt][c/96] * R[pt][c%96], scattered into the
    //      16x32 A-fragment layout with contiguous k (k = 16*half + e) ----
#pragma unroll
    for (int q = 0; q < 27; ++q) {
        int idx = t + 512 * q;                 // < 13824
        int pt = idx / 864, c = idx - pt * 864;
        int yy = c / 96, w = c - yy * 96;
        float v = sY[pt * 9 + yy] * sR[pt * 96 + w];
        int kc = c >> 5, rr = c & 31;
        int hf, e;
        hf = rr >> 4; e = rr & 15;                    // K 0..15 -> half0 e0..15, K 16..31 -> half1 e0..15
        int lane = (hf << 4) | pt;
        sA[(kc * 32 + lane) * 16 + e] = (_Float16)v;
    }
    __syncthreads();

    // ---- Phase B: per-wave 16x16 tile GEMM, K = 864 via 27 WMMAs ----
    const int wv = t >> 5, lane = t & 31;
    const v16h* __restrict__ bBase = reinterpret_cast<const v16h*>(mB);
    v8f acc = {};
#pragma unroll
    for (int kc = 0; kc < KC; ++kc) {
        v16h a = *reinterpret_cast<const v16h*>(&sA[(kc * 32 + lane) * 16]);
        v16h b = *(bBase + ((wv * KC + kc) * 32 + lane));
        if (kc + 1 < KC)
            __builtin_prefetch((const void*)(bBase + ((wv * KC + kc + 1) * 32 + lane)), 0, 1);
        acc = __builtin_amdgcn_wmma_f32_16x16x32_f16(
            /*neg_a=*/false, a, /*neg_b=*/false, b,
            /*c_mod=*/(short)0, acc, /*reuse_a=*/false, /*reuse_b=*/false);
    }

    // ---- Epilogue: D layout (VGPR j, lane L) -> M = j + 8*(L>=16), N = L&15 ----
    const int nc = lane & 15, hlf = lane >> 4;
    const int p  = wv * 16 + nc;
    const float k2v = k2[p];
#pragma unroll
    for (int j = 0; j < 8; ++j) {
        int M = j + hlf * 8;
        float v = sMask[M] ? acc[j] : k2v;
        out[(size_t)(n0 + M) * 256 + p] = v;
    }
}

extern "C" void kernel_launch(void* const* d_in, const int* in_sizes, int n_in,
                              void* d_out, int out_size, void* d_ws, size_t ws_size,
                              hipStream_t stream) {
    const float* r  = (const float*)d_in[0];   // [N,3]
    const float* M1 = (const float*)d_in[1];   // [16,16,9,96]
    const float* M2 = (const float*)d_in[2];   // [16,16,32]
    const float* wt = (const float*)d_in[3];   // [32]
    const float* W1 = (const float*)d_in[4];   // [1,64]
    const float* b1 = (const float*)d_in[5];   // [64]
    const float* W2 = (const float*)d_in[6];   // [64,96]
    const float* b2 = (const float*)d_in[7];   // [96]
    float* out = (float*)d_out;                // [N,16,16]

    _Float16* mB = (_Float16*)d_ws;                              // 432 KB
    float*    k2 = (float*)((char*)d_ws + MB_HALFS * sizeof(_Float16)); // +1 KB

    prep_mB<<<MB_HALFS / 512, 512, 0, stream>>>(M1, mB);
    prep_k2<<<1, 256, 0, stream>>>(M2, wt, k2);
    sh_mix_wmma<<<NPTS / 16, 512, 0, stream>>>(r, mB, k2, W1, b1, W2, b2, out);
}
